// MultiHeadedAttention_24206435680927
// MI455X (gfx1250) — matrix operand order repaired and checked by reading, not compiled, not run on hardware
//
#include <hip/hip_runtime.h>
#include <hip/hip_bf16.h>
#include <math.h>

// Problem constants (reference: B=2, S=2048, D=1024, H=16, DK=64)
#define BATCH  2
#define SEQ    2048
#define DMODEL 1024
#define NHEAD  16
#define DHEAD  64
#define MTOT   (BATCH * SEQ)        // 4096 rows for all projections

typedef __bf16 bf16;
typedef __attribute__((ext_vector_type(16))) __bf16 bf16x16;
typedef __attribute__((ext_vector_type(8)))  __bf16 bf16x8;
typedef __attribute__((ext_vector_type(8)))  float  f32x8;
typedef int b128i __attribute__((vector_size(16)));   // builtin's 128-bit unit

#define LOG2E 1.44269504088896340736f

// ---------------------------------------------------------------------------
// CDNA5 async global->LDS copy (ASYNCcnt path), with safe fallback.
// ---------------------------------------------------------------------------
#ifndef __has_builtin
#define __has_builtin(x) 0
#endif

#if __has_builtin(__builtin_amdgcn_global_load_async_to_lds_b128)
#define HAVE_ASYNC_LDS 1
#else
#define HAVE_ASYNC_LDS 0
#endif

// Copy 16 bytes (8 bf16) from global to LDS.
__device__ __forceinline__ void async_copy16(const bf16* g, bf16* l) {
#if HAVE_ASYNC_LDS
  __builtin_amdgcn_global_load_async_to_lds_b128(
      (__attribute__((address_space(1))) b128i*)(void*)g,
      (__attribute__((address_space(3))) b128i*)(void*)l,
      0, 0);
#else
  *(bf16x8*)l = *(const bf16x8*)g;
#endif
}

__device__ __forceinline__ void async_wait_all() {
#if HAVE_ASYNC_LDS
  asm volatile("s_wait_asynccnt 0" ::: "memory");
#endif
}

// ---------------------------------------------------------------------------
// WMMA helpers (CDNA5: V_WMMA_F32_16X16X32_BF16, wave32)
// ---------------------------------------------------------------------------
__device__ __forceinline__ f32x8 wmma_bf16f32(bf16x16 a, bf16x16 b, f32x8 c) {
  // (neg_a, A, neg_b, B, c_mod, C, reuse_a, reuse_b)
  return __builtin_amdgcn_wmma_f32_16x16x32_bf16(false, a, false, b, (short)0, c,
                                                 false, false);
}

__device__ __forceinline__ f32x8 zero8() {
  f32x8 z;
#pragma unroll
  for (int i = 0; i < 8; ++i) z[i] = 0.0f;
  return z;
}

// A-matrix fragment (16x32, bf16), from row-major [16][ld] tile.
// ISA layout: m = lane&15 (row), kh = lane>>4.
// elements 0..7  -> K = kh*8 + 0..7 ; elements 8..15 -> K = 16 + kh*8 + 0..7
__device__ __forceinline__ bf16x16 load_frag_a(const bf16* base, int ld) {
  const int lane = threadIdx.x & 31;
  const int m  = lane & 15;
  const int kh = lane >> 4;
  const bf16* p0 = base + (size_t)m * ld + kh * 8;
  bf16x8 lo = *(const bf16x8*)p0;
  bf16x8 hi = *(const bf16x8*)(p0 + 16);
  bf16x16 r;
#pragma unroll
  for (int i = 0; i < 8; ++i) { r[i] = lo[i]; r[i + 8] = hi[i]; }
  return r;
}

// B-matrix fragment (32x16, bf16), from tile stored as [N][ld] row-major
// (row n holds the K-contiguous column n of B).
// Column n=lane&15, kh=lane>>4: elements 0..7 -> K = kh*8 + 0..7 ; elements 8..15 -> K = 16 + kh*8 + 0..7
__device__ __forceinline__ bf16x16 load_frag_b(const bf16* base, int ld) {
  const int lane = threadIdx.x & 31;
  const int n  = lane & 15;
  const int k0 = (lane >> 4) * 8;
  const bf16* p = base + (size_t)n * ld + k0;
  bf16x8 lo = *(const bf16x8*)p;
  bf16x8 hi = *(const bf16x8*)(p + 16);
  bf16x16 r;
#pragma unroll
  for (int i = 0; i < 8; ++i) { r[i] = lo[i]; r[i + 8] = hi[i]; }
  return r;
}

// 16-lane-half reductions for C-matrix row statistics (wave32).
__device__ __forceinline__ float redmax16(float v) {
#pragma unroll
  for (int m = 1; m <= 8; m <<= 1) v = fmaxf(v, __shfl_xor(v, m, 32));
  return v;
}
__device__ __forceinline__ float redsum16(float v) {
#pragma unroll
  for (int m = 1; m <= 8; m <<= 1) v += __shfl_xor(v, m, 32);
  return v;
}

// ---------------------------------------------------------------------------
// fp32 -> bf16 cast
// ---------------------------------------------------------------------------
__global__ void cast_f32_bf16(const float* __restrict__ in, bf16* __restrict__ out,
                              int n) {
  int i = blockIdx.x * blockDim.x + threadIdx.x;
  if (i < n) out[i] = (bf16)in[i];
}

// ---------------------------------------------------------------------------
// Tiled WMMA GEMM:  C[M,N] = X[M,K] @ W[N,K]^T + bias[N]
// Block tile 128x128, BK=32, 256 threads = 8 waves in a 2(M) x 4(N) grid,
// each wave computes a 64x32 tile = 4x2 WMMA accumulators.
// Double-buffered LDS staging over the ASYNCcnt path.
// MODE 0: write bf16, head-split [B,H,S,DK].  MODE 1: write f32 row-major.
// Requires M%128==0, N%128==0, K%64==0 (holds here).
// ---------------------------------------------------------------------------
template <int MODE>
__global__ __launch_bounds__(256) void gemm_bf16_wmma(
    const bf16* __restrict__ X, const bf16* __restrict__ W,
    const float* __restrict__ bias, void* __restrict__ outv,
    int M, int N, int K) {
  __shared__ bf16 As[2][128][40];   // +8 bf16 pad: 16B-aligned rows, bank spread
  __shared__ bf16 Bs[2][128][40];

  const int tid  = threadIdx.x;
  const int wave = tid >> 5;
  const int wm   = wave >> 2;     // 0..1
  const int wn   = wave & 3;      // 0..3
  const int m0   = blockIdx.y * 128;
  const int n0   = blockIdx.x * 128;

  // Staging: 512 16B chunks per slab, 2 per thread.
  const int srow0 = tid >> 2;
  const int scol  = (tid & 3) * 8;

  f32x8 acc[4][2];
#pragma unroll
  for (int i = 0; i < 4; ++i)
#pragma unroll
    for (int j = 0; j < 2; ++j) acc[i][j] = zero8();

  // Prologue: stage k-slab 0 into buffer 0.
#pragma unroll
  for (int t = 0; t < 2; ++t) {
    int row = srow0 + t * 64;
    async_copy16(&X[(size_t)(m0 + row) * K + scol], &As[0][row][scol]);
    async_copy16(&W[(size_t)(n0 + row) * K + scol], &Bs[0][row][scol]);
  }
  async_wait_all();
  __syncthreads();

  for (int k0 = 0; k0 < K; k0 += 32) {
    const int cur = (k0 >> 5) & 1;
    // Issue async stage of the next k-slab into the other buffer.
    if (k0 + 32 < K) {
      const int nxt = cur ^ 1;
#pragma unroll
      for (int t = 0; t < 2; ++t) {
        int row = srow0 + t * 64;
        async_copy16(&X[(size_t)(m0 + row) * K + k0 + 32 + scol], &As[nxt][row][scol]);
        async_copy16(&W[(size_t)(n0 + row) * K + k0 + 32 + scol], &Bs[nxt][row][scol]);
      }
    }

    // Compute on the current buffer while the next slab is in flight.
    bf16x16 afr[4];
#pragma unroll
    for (int i = 0; i < 4; ++i)
      afr[i] = load_frag_a(&As[cur][wm * 64 + i * 16][0], 40);
#pragma unroll
    for (int j = 0; j < 2; ++j) {
      bf16x16 bfr = load_frag_b(&Bs[cur][wn * 32 + j * 16][0], 40);
#pragma unroll
      for (int i = 0; i < 4; ++i) acc[i][j] = wmma_bf16f32(afr[i], bfr, acc[i][j]);
    }

    async_wait_all();
    __syncthreads();
  }

  // Epilogue. C layout: lane holds col n = lane&15, rows e + 8*(lane>>4).
  const int lane = tid & 31;
  const int ln = lane & 15, kh = lane >> 4;
#pragma unroll
  for (int i = 0; i < 4; ++i) {
#pragma unroll
    for (int j = 0; j < 2; ++j) {
      int n = n0 + wn * 32 + j * 16 + ln;
      float bv = bias[n];
#pragma unroll
      for (int e = 0; e < 8; ++e) {
        int   m = m0 + wm * 64 + i * 16 + e + 8 * kh;
        float v = acc[i][j][e] + bv;
        if (MODE == 0) {
          int b = m >> 11;            // m / SEQ
          int s = m & (SEQ - 1);
          int h = n >> 6;             // n / DHEAD
          int dk = n & (DHEAD - 1);
          ((bf16*)outv)[((size_t)(b * NHEAD + h) * SEQ + s) * DHEAD + dk] = (bf16)v;
        } else {
          ((float*)outv)[(size_t)m * N + n] = v;
        }
      }
    }
  }
}

// ---------------------------------------------------------------------------
// Flash attention (per (b,h)): O = softmax(Q K^T / sqrt(DK) + mask) V
// Workgroup = 128 query rows (8 waves x 16 rows), streams 64-key tiles.
// K and raw-V tiles double-buffered via async global->LDS; V^T rebuilt in LDS.
// Q pre-scaled by 1/8 (exact in bf16). All GEMMs via WMMA bf16 -> f32.
// ---------------------------------------------------------------------------
#define FA_BM 128
#define FA_BN 64

__global__ __launch_bounds__(256) void flash_attn_wmma(
    const bf16* __restrict__ Qp, const bf16* __restrict__ Kp,
    const bf16* __restrict__ Vp, const int* __restrict__ mask,
    bf16* __restrict__ Out) {
  __shared__ bf16 Ks[2][FA_BN][72];     // K tile, [sk][dk]
  __shared__ bf16 Vr[2][FA_BN][72];     // raw V tile, [sk][dk]
  __shared__ bf16 Vt[DHEAD][72];        // V tile transposed, [dk][sk]
  __shared__ bf16 Ps[8][16][72];        // per-wave P staging (C->A relayout)

  const int tid  = threadIdx.x;
  const int wave = tid >> 5;
  const int lane = tid & 31;
  const int ln = lane & 15, kh = lane >> 4;

  const int bh  = blockIdx.y;
  const int b   = bh >> 4;              // / NHEAD
  const int h   = bh & (NHEAD - 1);
  const int qm0 = blockIdx.x * FA_BM + wave * 16;

  const bf16* Qb = Qp + (size_t)bh * SEQ * DHEAD;
  const bf16* Kb = Kp + (size_t)bh * SEQ * DHEAD;
  const bf16* Vb = Vp + (size_t)bh * SEQ * DHEAD;

  // Staging geometry: 512 16B chunks per tile, 2 per thread.
  const int trow0 = tid >> 3;           // 0..31 (+32 for t=1)
  const int tcol  = (tid & 7) * 8;      // dk chunk

  // Q fragments for the whole kernel, pre-scaled by 1/sqrt(DK) = 0.125.
  bf16x16 aq[2];
#pragma unroll
  for (int kk = 0; kk < 2; ++kk) {
    const bf16* p0 = Qb + (size_t)(qm0 + ln) * DHEAD + kk * 32 + kh * 8;
    bf16x8 lo = *(const bf16x8*)p0;
    bf16x8 hi = *(const bf16x8*)(p0 + 16);
#pragma unroll
    for (int e = 0; e < 8; ++e) {
      aq[kk][e]     = (bf16)((float)lo[e] * 0.125f);
      aq[kk][e + 8] = (bf16)((float)hi[e] * 0.125f);
    }
  }

  float m_s[8], l_s[8];
  f32x8 o[4];
#pragma unroll
  for (int j = 0; j < 8; ++j) { m_s[j] = -3.0e38f; l_s[j] = 0.0f; }
#pragma unroll
  for (int nt = 0; nt < 4; ++nt) o[nt] = zero8();

  // Prologue: stage K/V tile 0 into buffer 0.
#pragma unroll
  for (int t = 0; t < 2; ++t) {
    int row = trow0 + t * 32;
    async_copy16(&Kb[(size_t)row * DHEAD + tcol], &Ks[0][row][tcol]);
    async_copy16(&Vb[(size_t)row * DHEAD + tcol], &Vr[0][row][tcol]);
  }
  async_wait_all();
  __syncthreads();

  for (int kt = 0; kt < SEQ; kt += FA_BN) {
    const int cur = (kt >> 6) & 1;
    // Issue async stage of the next K/V tile.
    if (kt + FA_BN < SEQ) {
      const int nxt = cur ^ 1;
#pragma unroll
      for (int t = 0; t < 2; ++t) {
        int row = trow0 + t * 32;
        async_copy16(&Kb[(size_t)(kt + FA_BN + row) * DHEAD + tcol], &Ks[nxt][row][tcol]);
        async_copy16(&Vb[(size_t)(kt + FA_BN + row) * DHEAD + tcol], &Vr[nxt][row][tcol]);
      }
      __builtin_prefetch(&mask[((size_t)b * SEQ + qm0 + ln) * SEQ + kt + FA_BN], 0, 0);
    }

    // Build V^T for this tile from the raw staged copy (LDS->LDS transpose).
#pragma unroll
    for (int t = 0; t < 2; ++t) {
      int row = trow0 + t * 32;
      bf16x8 vv = *(const bf16x8*)&Vr[cur][row][tcol];
#pragma unroll
      for (int e = 0; e < 8; ++e) Vt[tcol + e][row] = vv[e];
    }
    __syncthreads();

    // S = Q K^T (already scaled): 16x64 scores = 4 n-tiles x 2 k-chunks.
    f32x8 s[4];
#pragma unroll
    for (int nt = 0; nt < 4; ++nt) {
      s[nt] = zero8();
#pragma unroll
      for (int kk = 0; kk < 2; ++kk) {
        bf16x16 bk = load_frag_b(&Ks[cur][nt * 16][kk * 32], 72);
        s[nt] = wmma_bf16f32(aq[kk], bk, s[nt]);
      }
    }

    // Mask + online softmax (row stats via 16-lane-half shuffles).
#pragma unroll
    for (int j = 0; j < 8; ++j) {
      int row = qm0 + j + 8 * kh;
      float mx = -3.0e38f;
#pragma unroll
      for (int nt = 0; nt < 4; ++nt) {
        int col = kt + nt * 16 + ln;
        float v = s[nt][j];
        if (mask[((size_t)b * SEQ + row) * SEQ + col] == 0) v = -1.0e9f;
        s[nt][j] = v;
        mx = fmaxf(mx, v);
      }
      mx = redmax16(mx);
      float mnew = fmaxf(m_s[j], mx);
      float corr = exp2f((m_s[j] - mnew) * LOG2E);
      float sum = 0.0f;
#pragma unroll
      for (int nt = 0; nt < 4; ++nt) {
        float p = exp2f((s[nt][j] - mnew) * LOG2E);
        s[nt][j] = p;
        sum += p;
      }
      sum = redsum16(sum);
      l_s[j] = l_s[j] * corr + sum;
      m_s[j] = mnew;
#pragma unroll
      for (int nt = 0; nt < 4; ++nt) o[nt][j] *= corr;
    }

    // P (C layout) -> LDS -> re-read as A fragments.
#pragma unroll
    for (int nt = 0; nt < 4; ++nt)
#pragma unroll
      for (int j = 0; j < 8; ++j)
        Ps[wave][j + 8 * kh][nt * 16 + ln] = (bf16)s[nt][j];

    // O += P @ V   (B = V^T tile, [dk][sk] in LDS)
#pragma unroll
    for (int kk = 0; kk < 2; ++kk) {
      bf16x16 ap = load_frag_a(&Ps[wave][0][kk * 32], 72);
#pragma unroll
      for (int nt = 0; nt < 4; ++nt) {
        bf16x16 bvf = load_frag_b(&Vt[nt * 16][kk * 32], 72);
        o[nt] = wmma_bf16f32(ap, bvf, o[nt]);
      }
    }

    async_wait_all();
    __syncthreads();
  }

  // O /= l, write bf16 to [B, S, H*DK] (input of output projection).
#pragma unroll
  for (int j = 0; j < 8; ++j) {
    float inv = 1.0f / l_s[j];
    int row = qm0 + j + 8 * kh;
#pragma unroll
    for (int nt = 0; nt < 4; ++nt) {
      int dk = nt * 16 + ln;
      Out[((size_t)b * SEQ + row) * DMODEL + h * DHEAD + dk] =
          (bf16)(o[nt][j] * inv);
    }
  }
}

// ---------------------------------------------------------------------------
// Host launcher
// ---------------------------------------------------------------------------
extern "C" void kernel_launch(void* const* d_in, const int* in_sizes, int n_in,
                              void* d_out, int out_size, void* d_ws,
                              size_t ws_size, hipStream_t stream) {
  (void)in_sizes; (void)n_in; (void)out_size; (void)ws_size;

  const float* q    = (const float*)d_in[0];
  const float* k    = (const float*)d_in[1];
  const float* v    = (const float*)d_in[2];
  const int*   mask = (const int*)d_in[3];
  const float* Wq   = (const float*)d_in[4];
  const float* bq   = (const float*)d_in[5];
  const float* Wk   = (const float*)d_in[6];
  const float* bk   = (const float*)d_in[7];
  const float* Wv   = (const float*)d_in[8];
  const float* bv   = (const float*)d_in[9];
  const float* Wo   = (const float*)d_in[10];
  const float* bo   = (const float*)d_in[11];

  const size_t XE = (size_t)MTOT * DMODEL;      // 4096*1024 elements
  const size_t WE = (size_t)DMODEL * DMODEL;

  char* ws = (char*)d_ws;
  size_t off = 0;
  auto carve = [&](size_t bytes) -> char* {
    char* p = ws + off;
    off += (bytes + 255) & ~(size_t)255;
    return p;
  };
  bf16* Xq  = (bf16*)carve(XE * 2);
  bf16* Xk  = (bf16*)carve(XE * 2);
  bf16* Xv  = (bf16*)carve(XE * 2);
  bf16* Wqb = (bf16*)carve(WE * 2);
  bf16* Wkb = (bf16*)carve(WE * 2);
  bf16* Wvb = (bf16*)carve(WE * 2);
  bf16* Wob = (bf16*)carve(WE * 2);
  bf16* Qp  = (bf16*)carve(XE * 2);   // [B,H,S,DK]
  bf16* Kp  = (bf16*)carve(XE * 2);
  bf16* Vp  = (bf16*)carve(XE * 2);
  bf16* Ao  = (bf16*)carve(XE * 2);   // attention output [B,S,D]

  // fp32 -> bf16 casts
  cast_f32_bf16<<<(int)(XE / 256), 256, 0, stream>>>(q, Xq, (int)XE);
  cast_f32_bf16<<<(int)(XE / 256), 256, 0, stream>>>(k, Xk, (int)XE);
  cast_f32_bf16<<<(int)(XE / 256), 256, 0, stream>>>(v, Xv, (int)XE);
  cast_f32_bf16<<<(int)(WE / 256), 256, 0, stream>>>(Wq, Wqb, (int)WE);
  cast_f32_bf16<<<(int)(WE / 256), 256, 0, stream>>>(Wk, Wkb, (int)WE);
  cast_f32_bf16<<<(int)(WE / 256), 256, 0, stream>>>(Wv, Wvb, (int)WE);
  cast_f32_bf16<<<(int)(WE / 256), 256, 0, stream>>>(Wo, Wob, (int)WE);

  // Q/K/V projections -> bf16 head-split [B,H,S,DK]
  dim3 ggrid(DMODEL / 128, MTOT / 128);  // (8, 32)
  gemm_bf16_wmma<0><<<ggrid, 256, 0, stream>>>(Xq, Wqb, bq, Qp, MTOT, DMODEL, DMODEL);
  gemm_bf16_wmma<0><<<ggrid, 256, 0, stream>>>(Xk, Wkb, bk, Kp, MTOT, DMODEL, DMODEL);
  gemm_bf16_wmma<0><<<ggrid, 256, 0, stream>>>(Xv, Wvb, bv, Vp, MTOT, DMODEL, DMODEL);

  // Flash attention
  dim3 fgrid(SEQ / FA_BM, BATCH * NHEAD);  // (16, 32)
  flash_attn_wmma<<<fgrid, 256, 0, stream>>>(Qp, Kp, Vp, mask, Ao);

  // Output projection -> fp32 d_out
  gemm_bf16_wmma<1><<<ggrid, 256, 0, stream>>>(Ao, Wob, bo, (float*)d_out,
                                               MTOT, DMODEL, DMODEL);
}
